// DecoderLayer_68040871903726
// MI455X (gfx1250) — compile-verified, matrix operand order checked by reading, not run on hardware
//
#include <hip/hip_runtime.h>
#include <hip/hip_bf16.h>

// ---------------------------------------------------------------------------
// MI455X (gfx1250) decoder layer.
// All matmuls -> v_wmma_f32_16x16x32_bf16 (f32 accumulate), f32 epilogues.
// Generic batched GEMM  C = alpha * A @ B^T (+bias) [+ReLU]:
//   workgroup tile 256x64, 8 waves of 64x32 (4x2 WMMA tiles each),
//   K staged in 32-wide slabs, double-buffered in LDS via the Tensor Data
//   Mover (tensor_load_to_lds + s_wait_tensorcnt), TDM pad feature produces
//   the 40-element padded LDS rows needed for conflict-friendly b128 reads.
// Fallback (no TDM builtin): manual global_load_b128 -> ds_store_b128 staging.
// ---------------------------------------------------------------------------

typedef __bf16 bf16_t;
typedef __attribute__((ext_vector_type(16))) __bf16 v16bf;
typedef __attribute__((ext_vector_type(8)))  float  v8f;
typedef __attribute__((ext_vector_type(4)))  unsigned int v4u;
typedef __attribute__((ext_vector_type(8)))  int v8i;
typedef __attribute__((ext_vector_type(4)))  int v4i;

constexpr long long cB  = 16;
constexpr long long cN  = 512;
constexpr long long cM  = 1024;
constexpr long long cH  = 512;   // HID
constexpr long long cNH = 8;
constexpr long long cHS = 64;

#define USE_TDM (__has_builtin(__builtin_amdgcn_tensor_load_to_lds) && \
                 __has_builtin(__builtin_amdgcn_s_wait_tensorcnt))

// ---------------------------------------------------------------------------
// fragment load: 16 bf16 per lane from an LDS row (K-contiguous), two b128
// reads; lane half selects the K sub-block per the 16-bit operand layout.
// ---------------------------------------------------------------------------
__device__ __forceinline__ v16bf ld_frag(const bf16_t* rowp, int half) {
  union { v16bf v; uint4 q[2]; } u;
  u.q[0] = *(const uint4*)(rowp + (half << 3));
  u.q[1] = *(const uint4*)(rowp + 16 + (half << 3));
  return u.v;
}

#if USE_TDM
// ---------------------------------------------------------------------------
// TDM 2D tile load: tile_dim0 = 32 elements (one K slab), tile_dim1 = rows.
// data_size = 2B. pad_enable with interval=16 DWORDs (one 64B row) and
// amount=4 DWORDs -> LDS row pitch 40 elements (80B), matching sA/sB.
// Descriptor D# per CDNA5 ISA 8.3/8.4; groups 2/3 zero (2D tensor).
// Generic->LDS pointer low 32 bits are the LDS byte offset.
// This toolchain's builtin takes 6 args (g0, g1, g2, g3, extra group, cpol).
// ---------------------------------------------------------------------------
__device__ __forceinline__ void tdm_load_2d(unsigned lds_addr, const bf16_t* gptr,
                                            unsigned tile_rows, long long row_stride) {
  const unsigned long long ga = (unsigned long long)(uintptr_t)gptr;
  const unsigned long long st = (unsigned long long)row_stride;  // elements
  const unsigned td0 = 0x40000000u;   // tensor_dim0: huge (tiles always in-bounds)
  const unsigned td1 = 0x00100000u;   // tensor_dim1: huge
  v4u g0;
  g0.x = 1u;                                            // count=1, user D#
  g0.y = lds_addr;                                      // LDS byte address
  g0.z = (unsigned)(ga & 0xFFFFFFFFu);                  // global_addr[31:0]
  g0.w = (unsigned)((ga >> 32) & 0x1FFFFFFu) | (2u << 30); // addr[56:32] | type=2
  v8i g1;
  g1[0] = (int)((1u << 16) | (1u << 20) | (3u << 22) | (3u << 25));
          // data_size=2B | pad_enable | pad_interval=16dw | pad_amount=4dw
  g1[1] = (int)((td0 & 0xFFFFu) << 16);                 // tensor_dim0[15:0]
  g1[2] = (int)(((td0 >> 16) & 0xFFFFu) | ((td1 & 0xFFFFu) << 16));
  g1[3] = (int)(((td1 >> 16) & 0xFFFFu) | (32u << 16)); // tile_dim0 = 32
  g1[4] = (int)(tile_rows & 0xFFFFu);                   // tile_dim1, tile_dim2=0
  g1[5] = (int)(st & 0xFFFFFFFFu);                      // tensor_dim0_stride lo
  g1[6] = (int)((st >> 32) & 0xFFFFu);                  // stride hi | dim1_stride lo
  g1[7] = 0;
  v4i z4 = {};
  v8i z8 = {};
  __builtin_amdgcn_tensor_load_to_lds(g0, g1, z4, z4, z8, 0);
}
#endif

template<int OUT_BF16, int RELU>
__global__ __launch_bounds__(256)
void k_gemm_bt(const bf16_t* __restrict__ A, long long a_rs, long long a_bs,
               const bf16_t* __restrict__ Bw, long long b_rs, long long b_bs,
               void* __restrict__ Cv, long long c_rs, long long c_bs,
               const float* __restrict__ bias, float alpha, int K)
{
  __shared__ __align__(16) bf16_t sA[2][256][40];  // 256x32 used + pad
  __shared__ __align__(16) bf16_t sB[2][64][40];

  const int t    = threadIdx.x;
  const int lane = t & 31;
  const int wave = t >> 5;          // 0..7
  const int half = lane >> 4;       // 0/1
  const int idx  = lane & 15;
  const int wm   = wave >> 1;       // 0..3 -> M offset wm*64
  const int wn   = wave & 1;        // 0..1 -> N offset wn*32

  const long long tileM = (long long)blockIdx.y * 256;
  const long long tileN = (long long)blockIdx.x * 64;
  const long long bz    = blockIdx.z;

  const bf16_t* Ab = A  + bz * a_bs + tileM * a_rs;
  const bf16_t* Bb = Bw + bz * b_bs + tileN * b_rs;

  v8f zero = {};
  v8f acc[4][2];
  #pragma unroll
  for (int i = 0; i < 4; ++i) { acc[i][0] = zero; acc[i][1] = zero; }

#if USE_TDM
  const bool issuer = (wave == 0);
  if (issuer) {
    tdm_load_2d((unsigned)(uintptr_t)&sA[0][0][0], Ab, 256, a_rs);
    tdm_load_2d((unsigned)(uintptr_t)&sB[0][0][0], Bb, 64,  b_rs);
  }
#else
  // manual staging map: A = 1024 16B chunks (4/thread), B = 256 (1/thread)
  const int akc = (t & 3) << 3;     // k offset 0/8/16/24
  const int br  = t >> 2;
  const int bkc = akc;
#endif

  int buf = 0;
  for (int k0 = 0; k0 < K; k0 += 32, buf ^= 1) {
#if USE_TDM
    if (issuer) {
      if (k0 + 32 < K) {
        tdm_load_2d((unsigned)(uintptr_t)&sA[buf ^ 1][0][0], Ab + k0 + 32, 256, a_rs);
        tdm_load_2d((unsigned)(uintptr_t)&sB[buf ^ 1][0][0], Bb + k0 + 32, 64,  b_rs);
        __builtin_amdgcn_s_wait_tensorcnt((short)2);   // current slab's pair done
      } else {
        __builtin_amdgcn_s_wait_tensorcnt((short)0);
      }
    }
    __syncthreads();
#else
    #pragma unroll
    for (int c = 0; c < 4; ++c) {
      const int r = (t + c * 256) >> 2;
      *(uint4*)&sA[buf][r][akc] = *(const uint4*)(Ab + (long long)r * a_rs + k0 + akc);
    }
    *(uint4*)&sB[buf][br][bkc] = *(const uint4*)(Bb + (long long)br * b_rs + k0 + bkc);
    __syncthreads();
#endif

    v16bf af[4], bfr[2];
    #pragma unroll
    for (int i = 0; i < 4; ++i)
      af[i] = ld_frag(&sA[buf][wm * 64 + i * 16 + idx][0], half);
    #pragma unroll
    for (int j = 0; j < 2; ++j)
      bfr[j] = ld_frag(&sB[buf][wn * 32 + j * 16 + idx][0], half);

    #pragma unroll
    for (int i = 0; i < 4; ++i) {
      acc[i][0] = __builtin_amdgcn_wmma_f32_16x16x32_bf16(false, af[i], false, bfr[0], (short)0, acc[i][0], false, false);
      acc[i][1] = __builtin_amdgcn_wmma_f32_16x16x32_bf16(false, af[i], false, bfr[1], (short)0, acc[i][1], false, false);
    }
    __syncthreads();
  }

  float*  Cf = (float*)Cv;
  bf16_t* Cb = (bf16_t*)Cv;
  #pragma unroll
  for (int mi = 0; mi < 4; ++mi) {
    #pragma unroll
    for (int ni = 0; ni < 2; ++ni) {
      const long long col = tileN + wn * 32 + ni * 16 + idx;
      const float bv = bias ? bias[col] : 0.0f;
      #pragma unroll
      for (int r = 0; r < 8; ++r) {
        const long long row = tileM + wm * 64 + mi * 16 + half * 8 + r;
        float v = acc[mi][ni][r] * alpha + bv;
        if (RELU) v = fmaxf(v, 0.0f);
        const long long off = bz * c_bs + row * c_rs + col;
        if (OUT_BF16) Cb[off] = (bf16_t)v; else Cf[off] = v;
      }
    }
  }
}

// ---------------------------------------------------------------------------
// elementwise / reduction kernels (f32 math, bf16 stores for GEMM operands)
// ---------------------------------------------------------------------------
__global__ void k_cvt_bf16(const float* __restrict__ x, bf16_t* __restrict__ y,
                           long long n) {
  long long i  = (long long)blockIdx.x * blockDim.x + threadIdx.x;
  long long st = (long long)gridDim.x * blockDim.x;
  for (; i < n; i += st) y[i] = (bf16_t)x[i];
}

// qkv f32 (B,N,NH,3*HS) -> qh/kh (B,NH,N,HS) bf16, vhT (B,NH,HS,N) bf16
__global__ void k_split_qkv(const float* __restrict__ qkv,
                            bf16_t* __restrict__ qh, bf16_t* __restrict__ kh,
                            bf16_t* __restrict__ vhT) {
  const long long n_tot = cB * cN * cNH * cHS;
  long long i  = (long long)blockIdx.x * blockDim.x + threadIdx.x;
  long long st = (long long)gridDim.x * blockDim.x;
  for (; i < n_tot; i += st) {
    const int  d = (int)(i & 63);
    long long  t = i >> 6;
    const int  h = (int)(t & 7);  t >>= 3;
    const long long n = t % cN;
    const long long b = t / cN;
    const long long base = (b * cN + n) * (3 * cH) + h * (3 * cHS);
    const float q = qkv[base + d];
    const float k = qkv[base + cHS + d];
    const float v = qkv[base + 2 * cHS + d];
    const long long bh = b * cNH + h;
    qh [(bh * cN + n) * cHS + d] = (bf16_t)q;
    kh [(bh * cN + n) * cHS + d] = (bf16_t)k;
    vhT[(bh * cHS + d) * cN + n] = (bf16_t)v;
  }
}

// q f32 (B,N,HID) + sin/cos (B,N,HS/2) -> rotated qh (B,NH,N,HS) bf16
__global__ void k_q_moverz(const float* __restrict__ qf,
                           const float* __restrict__ sn, const float* __restrict__ cs,
                           bf16_t* __restrict__ qh) {
  const long long n_tot = cB * cN * cNH * (cHS / 2);
  long long i  = (long long)blockIdx.x * blockDim.x + threadIdx.x;
  long long st = (long long)gridDim.x * blockDim.x;
  for (; i < n_tot; i += st) {
    const int  d = (int)(i & 31);
    long long  t = i >> 5;
    const int  h = (int)(t & 7);  t >>= 3;
    const long long n = t % cN;
    const long long b = t / cN;
    const long long row = b * cN + n;
    const float x0 = qf[row * cH + h * cHS + 2 * d];
    const float x1 = qf[row * cH + h * cHS + 2 * d + 1];
    const float s  = sn[row * (cHS / 2) + d];
    const float c  = cs[row * (cHS / 2) + d];
    const long long bh = b * cNH + h;
    qh[(bh * cN + n) * cHS + d]            = (bf16_t)(x0 * c - x1 * s);
    qh[(bh * cN + n) * cHS + cHS / 2 + d]  = (bf16_t)(x1 * c + x0 * s);
  }
}

// kv f32 (B,M,NH,2*HS) + sin/cos (B,M,HS/2) -> rotated kh (B,NH,M,HS),
// vhT (B,NH,HS,M) (plain, transposed) as bf16
__global__ void k_kv_moverz(const float* __restrict__ kvf,
                            const float* __restrict__ sn, const float* __restrict__ cs,
                            bf16_t* __restrict__ kh, bf16_t* __restrict__ vhT) {
  const long long n_tot = cB * cM * cNH * (cHS / 2);
  long long i  = (long long)blockIdx.x * blockDim.x + threadIdx.x;
  long long st = (long long)gridDim.x * blockDim.x;
  for (; i < n_tot; i += st) {
    const int  d = (int)(i & 31);
    long long  t = i >> 5;
    const int  h = (int)(t & 7);  t >>= 3;
    const long long m = t % cM;
    const long long b = t / cM;
    const long long row  = b * cM + m;
    const long long base = row * (2 * cH) + h * (2 * cHS);
    const float k0 = kvf[base + 2 * d];
    const float k1 = kvf[base + 2 * d + 1];
    const float s  = sn[row * (cHS / 2) + d];
    const float c  = cs[row * (cHS / 2) + d];
    const long long bh = b * cNH + h;
    kh[(bh * cM + m) * cHS + d]           = (bf16_t)(k0 * c - k1 * s);
    kh[(bh * cM + m) * cHS + cHS / 2 + d] = (bf16_t)(k1 * c + k0 * s);
    const float v0 = kvf[base + cHS + 2 * d];
    const float v1 = kvf[base + cHS + 2 * d + 1];
    vhT[(bh * cHS + 2 * d)     * cM + m] = (bf16_t)v0;
    vhT[(bh * cHS + 2 * d + 1) * cM + m] = (bf16_t)v1;
  }
}

// av f32 (B,NH,N,HS) -> merged bf16 (B,N,HID)
__global__ void k_merge_heads(const float* __restrict__ av, bf16_t* __restrict__ xm) {
  const long long n_tot = cB * cN * cH;
  long long i  = (long long)blockIdx.x * blockDim.x + threadIdx.x;
  long long st = (long long)gridDim.x * blockDim.x;
  for (; i < n_tot; i += st) {
    const int  d = (int)(i % cH);
    long long  t = i / cH;
    const long long n = t % cN;
    const long long b = t / cN;
    const int h  = d >> 6;
    const int hd = d & 63;
    xm[i] = (bf16_t)av[((b * cNH + h) * cN + n) * cHS + hd];
  }
}

// one block per score row; masked softmax, bf16 probabilities out
__global__ __launch_bounds__(256)
void k_softmax(const float* __restrict__ S, bf16_t* __restrict__ P,
               int L, int qlen, int causal, int pad_start) {
  const long long row = blockIdx.x;
  const int n = (int)(row % qlen);
  const float*  s = S + row * (long long)L;
  bf16_t*       p = P + row * (long long)L;
  __shared__ float red[256];
  const int t = threadIdx.x;

  float mx = -INFINITY;
  for (int j = t; j < L; j += 256) {
    const bool masked = (causal && j > n) || (pad_start >= 0 && j >= pad_start);
    mx = fmaxf(mx, masked ? -INFINITY : s[j]);
  }
  red[t] = mx; __syncthreads();
  for (int o = 128; o > 0; o >>= 1) { if (t < o) red[t] = fmaxf(red[t], red[t + o]); __syncthreads(); }
  mx = red[0]; __syncthreads();

  float sum = 0.0f;
  for (int j = t; j < L; j += 256) {
    const bool masked = (causal && j > n) || (pad_start >= 0 && j >= pad_start);
    sum += masked ? 0.0f : __expf(s[j] - mx);
  }
  red[t] = sum; __syncthreads();
  for (int o = 128; o > 0; o >>= 1) { if (t < o) red[t] += red[t + o]; __syncthreads(); }
  const float inv = 1.0f / red[0];

  for (int j = t; j < L; j += 256) {
    const bool masked = (causal && j > n) || (pad_start >= 0 && j >= pad_start);
    p[j] = (bf16_t)(masked ? 0.0f : __expf(s[j] - mx) * inv);
  }
}

// y = LN(X + R) * g + b ; writes f32 (residual chain / output) and optional bf16
__global__ __launch_bounds__(256)
void k_add_ln(const float* __restrict__ X, const float* __restrict__ R,
              const float* __restrict__ g, const float* __restrict__ bta,
              float* __restrict__ of, bf16_t* __restrict__ ob) {
  const long long base = (long long)blockIdx.x * cH;
  const int t = threadIdx.x;
  __shared__ float s1[256], s2[256];
  const float a0 = X[base + t]       + R[base + t];
  const float a1 = X[base + t + 256] + R[base + t + 256];
  s1[t] = a0 + a1;
  s2[t] = a0 * a0 + a1 * a1;
  __syncthreads();
  for (int o = 128; o > 0; o >>= 1) {
    if (t < o) { s1[t] += s1[t + o]; s2[t] += s2[t + o]; }
    __syncthreads();
  }
  const float mean = s1[0] * (1.0f / (float)cH);
  const float var  = s2[0] * (1.0f / (float)cH) - mean * mean;
  const float rstd = rsqrtf(var + 1e-5f);
  const float o0 = (a0 - mean) * rstd * g[t]       + bta[t];
  const float o1 = (a1 - mean) * rstd * g[t + 256] + bta[t + 256];
  of[base + t]       = o0;
  of[base + t + 256] = o1;
  if (ob) { ob[base + t] = (bf16_t)o0; ob[base + t + 256] = (bf16_t)o1; }
}

// ---------------------------------------------------------------------------
// workspace layout (bytes). Arena R (268MB) / P (134MB) are reused across
// pipeline stages with strictly disjoint lifetimes. Total ~528 MB (504 MiB).
// ---------------------------------------------------------------------------
constexpr size_t OFF_TGT_BF = 0;                                   // 8 MB
constexpr size_t OFF_MEM_BF = OFF_TGT_BF + 8388608;                // 16 MB
constexpr size_t OFF_W_QKV  = OFF_MEM_BF + 16777216;               // 1.5 MB
constexpr size_t OFF_W_OW1  = OFF_W_QKV  + 1572864;                // 0.5 MB
constexpr size_t OFF_W_QW   = OFF_W_OW1  + 524288;                 // 0.5 MB
constexpr size_t OFF_W_KVW  = OFF_W_QW   + 524288;                 // 1 MB
constexpr size_t OFF_W_OW2  = OFF_W_KVW  + 1048576;                // 0.5 MB
constexpr size_t OFF_W_FF1  = OFF_W_OW2  + 524288;                 // 2 MB
constexpr size_t OFF_W_FF2  = OFF_W_FF1  + 2097152;                // 2 MB
constexpr size_t OFF_QH     = OFF_W_FF2  + 2097152;                // 8 MB
constexpr size_t OFF_KH     = OFF_QH     + 8388608;                // 16 MB (max L=M)
constexpr size_t OFF_VHT    = OFF_KH     + 16777216;               // 16 MB
constexpr size_t OFF_T1F    = OFF_VHT    + 16777216;               // 16 MB
constexpr size_t OFF_T1B    = OFF_T1F    + 16777216;               // 8 MB
constexpr size_t OFF_T2F    = OFF_T1B    + 8388608;                // 16 MB
constexpr size_t OFF_T2B    = OFF_T2F    + 16777216;               // 8 MB
constexpr size_t OFF_R      = OFF_T2B    + 8388608;                // 268 MB arena
constexpr size_t OFF_P      = OFF_R      + 268435456;              // 134 MB arena

extern "C" void kernel_launch(void* const* d_in, const int* in_sizes, int n_in,
                              void* d_out, int out_size, void* d_ws, size_t ws_size,
                              hipStream_t stream) {
  (void)in_sizes; (void)n_in; (void)out_size; (void)ws_size;

  const float* tgt      = (const float*)d_in[0];
  const float* mem      = (const float*)d_in[1];
  const float* psin     = (const float*)d_in[2];
  const float* pcos     = (const float*)d_in[3];
  const float* ksin     = (const float*)d_in[4];
  const float* kcos     = (const float*)d_in[5];
  const float* mmha_w   = (const float*)d_in[8];
  const float* mmha_b   = (const float*)d_in[9];
  const float* mmha_ow  = (const float*)d_in[10];
  const float* mmha_ob  = (const float*)d_in[11];
  const float* mmha_g   = (const float*)d_in[12];
  const float* mmha_bt  = (const float*)d_in[13];
  const float* mha_qw   = (const float*)d_in[14];
  const float* mha_qb   = (const float*)d_in[15];
  const float* mha_kvw  = (const float*)d_in[16];
  const float* mha_kvb  = (const float*)d_in[17];
  const float* mha_ow   = (const float*)d_in[18];
  const float* mha_ob   = (const float*)d_in[19];
  const float* mha_g    = (const float*)d_in[20];
  const float* mha_bt   = (const float*)d_in[21];
  const float* ffn_w1   = (const float*)d_in[22];
  const float* ffn_w2   = (const float*)d_in[23];
  const float* ffn_g    = (const float*)d_in[24];
  const float* ffn_bt   = (const float*)d_in[25];

  char* ws = (char*)d_ws;
  bf16_t* tgt_bf = (bf16_t*)(ws + OFF_TGT_BF);
  bf16_t* mem_bf = (bf16_t*)(ws + OFF_MEM_BF);
  bf16_t* w_qkv  = (bf16_t*)(ws + OFF_W_QKV);
  bf16_t* w_ow1  = (bf16_t*)(ws + OFF_W_OW1);
  bf16_t* w_qw   = (bf16_t*)(ws + OFF_W_QW);
  bf16_t* w_kvw  = (bf16_t*)(ws + OFF_W_KVW);
  bf16_t* w_ow2  = (bf16_t*)(ws + OFF_W_OW2);
  bf16_t* w_ff1  = (bf16_t*)(ws + OFF_W_FF1);
  bf16_t* w_ff2  = (bf16_t*)(ws + OFF_W_FF2);
  bf16_t* qh     = (bf16_t*)(ws + OFF_QH);
  bf16_t* kh     = (bf16_t*)(ws + OFF_KH);
  bf16_t* vhT    = (bf16_t*)(ws + OFF_VHT);
  float*  t1f    = (float*)(ws + OFF_T1F);
  bf16_t* t1b    = (bf16_t*)(ws + OFF_T1B);
  float*  t2f    = (float*)(ws + OFF_T2F);
  bf16_t* t2b    = (bf16_t*)(ws + OFF_T2B);

  // arena R sub-buffers (lifetimes strictly sequential)
  float*  qkvf   = (float*)(ws + OFF_R);                    // 48 MB
  float*  scores = (float*)(ws + OFF_R);                    // up to 268 MB
  float*  av     = (float*)(ws + OFF_R);                    // 16 MB
  bf16_t* xm     = (bf16_t*)(ws + OFF_R + 16777216);        // 8 MB
  float*  proj   = (float*)(ws + OFF_R + 25165824);         // 16 MB
  float*  qpf    = (float*)(ws + OFF_R);                    // 16 MB (cross prep)
  float*  kvf    = (float*)(ws + OFF_R + 16777216);         // 64 MB (cross prep)
  bf16_t* h1b    = (bf16_t*)(ws + OFF_R);                   // 32 MB (FFN)
  float*  h2f    = (float*)(ws + OFF_R + 33554432);         // 16 MB (FFN)
  bf16_t* probs  = (bf16_t*)(ws + OFF_P);                   // up to 134 MB

  auto cvt = [&](const float* src, bf16_t* dst, long long n) {
    int blocks = (int)((n + 1023) / 1024);
    k_cvt_bf16<<<blocks, 256, 0, stream>>>(src, dst, n);
  };
  auto gemm = [&](const bf16_t* A, long long a_rs, long long a_bs,
                  const bf16_t* Bp, long long b_rs, long long b_bs,
                  float* C, long long c_rs, long long c_bs,
                  const float* bias, float alpha,
                  int Mrows, int Nout, int K, int batch) {
    dim3 g(Nout / 64, Mrows / 256, batch);
    k_gemm_bt<0, 0><<<g, 256, 0, stream>>>(A, a_rs, a_bs, Bp, b_rs, b_bs,
                                           (void*)C, c_rs, c_bs, bias, alpha, K);
  };

  const long long BN = cB * cN;   // 8192
  const long long BM = cB * cM;   // 16384

  // ---- precision conversion (activations + weights -> bf16) ----
  cvt(tgt,     tgt_bf, BN * cH);
  cvt(mem,     mem_bf, BM * cH);
  cvt(mmha_w,  w_qkv,  3 * cH * cH);
  cvt(mmha_ow, w_ow1,  cH * cH);
  cvt(mha_qw,  w_qw,   cH * cH);
  cvt(mha_kvw, w_kvw,  2 * cH * cH);
  cvt(mha_ow,  w_ow2,  cH * cH);
  cvt(ffn_w1,  w_ff1,  4 * cH * cH);
  cvt(ffn_w2,  w_ff2,  cH * 4 * cH);

  // ---- masked self-attention ----
  gemm(tgt_bf, cH, 0, w_qkv, cH, 0, qkvf, 3 * cH, 0, mmha_b, 1.0f,
       (int)BN, (int)(3 * cH), (int)cH, 1);
  k_split_qkv<<<(int)((cB * cN * cNH * cHS + 1023) / 1024), 256, 0, stream>>>(qkvf, qh, kh, vhT);
  gemm(qh, cHS, cN * cHS, kh, cHS, cN * cHS, scores, cN, cN * cN,
       nullptr, 0.125f, (int)cN, (int)cN, (int)cHS, (int)(cB * cNH));
  k_softmax<<<(int)(cB * cNH * cN), 256, 0, stream>>>(scores, probs, (int)cN, (int)cN, 1, -1);
  gemm((const bf16_t*)probs, cN, cN * cN, vhT, cN, cHS * cN, av, cHS, cN * cHS,
       nullptr, 1.0f, (int)cN, (int)cHS, (int)cN, (int)(cB * cNH));
  k_merge_heads<<<(int)((BN * cH + 1023) / 1024), 256, 0, stream>>>(av, xm);
  gemm(xm, cH, 0, w_ow1, cH, 0, proj, cH, 0, mmha_ob, 1.0f, (int)BN, (int)cH, (int)cH, 1);
  k_add_ln<<<(int)BN, 256, 0, stream>>>(proj, tgt, mmha_g, mmha_bt, t1f, t1b);

  // ---- cross-attention with moverz rotation ----
  gemm(t1b, cH, 0, w_qw, cH, 0, qpf, cH, 0, mha_qb, 1.0f, (int)BN, (int)cH, (int)cH, 1);
  gemm(mem_bf, cH, 0, w_kvw, cH, 0, kvf, 2 * cH, 0, mha_kvb, 1.0f,
       (int)BM, (int)(2 * cH), (int)cH, 1);
  k_q_moverz<<<(int)((cB * cN * cNH * 32 + 1023) / 1024), 256, 0, stream>>>(qpf, psin, pcos, qh);
  k_kv_moverz<<<(int)((cB * cM * cNH * 32 + 1023) / 1024), 256, 0, stream>>>(kvf, ksin, kcos, kh, vhT);
  gemm(qh, cHS, cN * cHS, kh, cHS, cM * cHS, scores, cM, cN * cM,
       nullptr, 0.125f, (int)cN, (int)cM, (int)cHS, (int)(cB * cNH));
  k_softmax<<<(int)(cB * cNH * cN), 256, 0, stream>>>(scores, probs, (int)cM, (int)cN, 0, (int)(cM - 64));
  gemm((const bf16_t*)probs, cM, cN * cM, vhT, cM, cHS * cM, av, cHS, cN * cHS,
       nullptr, 1.0f, (int)cN, (int)cHS, (int)cM, (int)(cB * cNH));
  k_merge_heads<<<(int)((BN * cH + 1023) / 1024), 256, 0, stream>>>(av, xm);
  gemm(xm, cH, 0, w_ow2, cH, 0, proj, cH, 0, mha_ob, 1.0f, (int)BN, (int)cH, (int)cH, 1);
  k_add_ln<<<(int)BN, 256, 0, stream>>>(proj, t1f, mha_g, mha_bt, t2f, t2b);

  // ---- FFN (ReLU fused into GEMM1 epilogue, bf16 out) ----
  {
    dim3 g((int)(4 * cH / 64), (int)(BN / 256), 1);
    k_gemm_bt<1, 1><<<g, 256, 0, stream>>>(t2b, cH, 0, w_ff1, cH, 0,
                                           (void*)h1b, 4 * cH, 0, nullptr, 1.0f, (int)cH);
  }
  gemm(h1b, 4 * cH, 0, w_ff2, 4 * cH, 0, h2f, cH, 0, nullptr, 1.0f,
       (int)BN, (int)cH, (int)(4 * cH), 1);
  k_add_ln<<<(int)BN, 256, 0, stream>>>(h2f, t2f, ffn_g, ffn_bt, (float*)d_out, nullptr);
}
